// Block2DGRU_12532714570189
// MI455X (gfx1250) — compile-verified, matrix operand order checked by reading, not run on hardware
//
#include <hip/hip_runtime.h>
#include <hip/hip_bf16.h>

// ---------------- types / constants ----------------
typedef __bf16 bf16;
typedef __attribute__((ext_vector_type(16))) __bf16 v16bf;
typedef __attribute__((ext_vector_type(8)))  float  v8f;
typedef __attribute__((ext_vector_type(4)))  int    v4i_;

#define DIM_  384
#define DI_   768          // 2*DIM
#define HGC   1536         // 2*DI
#define MLP_  1536         // 4*DIM
#define NB    32
#define LL    1024         // H*W
#define MTOT  (NB*LL)      // 32768 token rows
#define EPS_  1e-5f

#define BM 128
#define BN 64
#define BK 32

// ---------------- CDNA5 async global->LDS helpers ----------------
typedef __attribute__((address_space(1))) v4i_ gv4;   // global int4
typedef __attribute__((address_space(3))) v4i_ lv4;   // LDS int4

__device__ __forceinline__ void async_copy_b128(const bf16* g, bf16* l) {
#if __has_builtin(__builtin_amdgcn_global_load_async_to_lds_b128)
    __builtin_amdgcn_global_load_async_to_lds_b128(
        (gv4*)(void*)const_cast<bf16*>(g),
        (lv4*)(void*)l, 0, 0);
#else
    asm volatile("global_load_async_to_lds_b128 %0, %1, off"
                 :: "v"((unsigned)(unsigned long)(lv4*)(void*)l),
                    "v"((gv4*)(void*)const_cast<bf16*>(g))
                 : "memory");
#endif
}
__device__ __forceinline__ void wait_async0() {
#if __has_builtin(__builtin_amdgcn_s_wait_asynccnt)
    __builtin_amdgcn_s_wait_asynccnt(0);
#else
    asm volatile("s_wait_asynccnt 0" ::: "memory");
#endif
}

// ---------------- math helpers ----------------
__device__ __forceinline__ float softplusf_(float x) {
    return fmaxf(x, 0.f) + log1pf(expf(-fabsf(x)));
}
__device__ __forceinline__ float logaddexpf_(float a, float b) {
    float m = fmaxf(a, b);
    return m + log1pf(expf(fminf(a, b) - m));
}
__device__ __forceinline__ float gelu_exact(float x) {
    return 0.5f * x * (1.f + erff(x * 0.70710678118654752440f));
}

// ---------------- small kernels ----------------
__global__ void cvt_bf16_kernel(const float* __restrict__ s, bf16* __restrict__ d, int n) {
    int i = blockIdx.x * 256 + threadIdx.x;
    if (i < n) d[i] = (bf16)s[i];
}

template <typename OutT>
__global__ __launch_bounds__(256) void ln_kernel(const float* __restrict__ x,
                                                 const float* __restrict__ g,
                                                 const float* __restrict__ b,
                                                 OutT* __restrict__ out) {
    int gw   = (blockIdx.x * blockDim.x + threadIdx.x) >> 5;  // wave per row
    int lane = threadIdx.x & 31;
    if (gw >= MTOT) return;
    const float* row = x + (size_t)gw * DIM_;
    float vals[DIM_ / 32];
    float s = 0.f, s2 = 0.f;
#pragma unroll
    for (int i = 0; i < DIM_ / 32; ++i) {
        float v = row[lane + 32 * i];
        vals[i] = v; s += v; s2 += v * v;
    }
#pragma unroll
    for (int o = 16; o > 0; o >>= 1) {
        s  += __shfl_xor(s, o, 32);
        s2 += __shfl_xor(s2, o, 32);
    }
    float mu  = s / DIM_;
    float var = s2 / DIM_ - mu * mu;
    float inv = rsqrtf(var + EPS_);
#pragma unroll
    for (int i = 0; i < DIM_ / 32; ++i) {
        int c = lane + 32 * i;
        out[(size_t)gw * DIM_ + c] = (OutT)((vals[i] - mu) * inv * g[c] + b[c]);
    }
}

__global__ __launch_bounds__(256) void dwconv_kernel(const float* __restrict__ xn,
                                                     const float* __restrict__ w,
                                                     const float* __restrict__ bias,
                                                     bf16* __restrict__ xs) {
    size_t idx = (size_t)blockIdx.x * 256 + threadIdx.x;
    int c   = (int)(idx % DIM_);
    size_t r = idx / DIM_;
    int xx  = (int)(r % 32); r /= 32;
    int yy  = (int)(r % 32);
    int n   = (int)(r / 32);
    float acc = bias[c];
#pragma unroll
    for (int dy = 0; dy < 3; ++dy) {
#pragma unroll
        for (int dx = 0; dx < 3; ++dx) {
            int iy = yy + dy - 1, ix = xx + dx - 1;
            if (iy >= 0 && iy < 32 && ix >= 0 && ix < 32)
                acc += xn[(((size_t)n * 32 + iy) * 32 + ix) * DIM_ + c] * w[c * 9 + dy * 3 + dx];
        }
    }
    xs[idx] = (bf16)acc;
}

// sequential log-space minGRU scan; one thread per (batch, channel)
__global__ __launch_bounds__(256) void scan_kernel(const bf16* __restrict__ hg,
                                                   bf16* __restrict__ h) {
    int idx = blockIdx.x * 256 + threadIdx.x;
    if (idx >= NB * DI_) return;
    int n = idx / DI_, c = idx % DI_;
    const bf16* base = hg + (size_t)n * LL * HGC;
    bf16* out = h + (size_t)n * LL * DI_ + c;
    float lh = 0.f;
    for (int t = 0; t < LL; ++t) {
        float hid  = (float)base[(size_t)t * HGC + c];
        float gate = (float)base[(size_t)t * HGC + DI_ + c];
        float lc = -softplusf_(gate);                      // log sigmoid(-gate)
        float lg = (hid >= 0.f) ? logf(hid + 0.5f) : -softplusf_(-hid);
        float lv = -softplusf_(-gate) + lg;                // log sigmoid(gate) + log g(hid)
        lh = (t == 0) ? lv : logaddexpf_(lc + lh, lv);
        out[(size_t)t * DI_] = (bf16)expf(lh);
    }
}

// ---------------- WMMA fragment helpers ----------------
// A fragment (16x32 bf16): lane L holds row M=L&15; K = (L>>4)*8 + [0..7] and 16+(L>>4)*8+[0..7]
__device__ __forceinline__ v16bf frag16(const bf16* rowp, int khalf) {
    union { uint4 u[2]; v16bf v; } t;
    t.u[0] = *reinterpret_cast<const uint4*>(rowp + khalf * 8);
    t.u[1] = *reinterpret_cast<const uint4*>(rowp + 16 + khalf * 8);
    return t.v;
}

// ---------------- GEMM: C[MxN] = A_bf16[MxK] x B_bf16[KxN] ----------------
// EPI 0: store bf16 raw; 1: bias+GELU -> bf16; 2: bias+residual -> f32
template <int EPI, bool REV>
__global__ __launch_bounds__(256) void gemm_kernel(const bf16* __restrict__ A,
                                                   const bf16* __restrict__ B,
                                                   const float* __restrict__ bias,
                                                   const float* __restrict__ res,
                                                   void* __restrict__ C,
                                                   int M, int Ntot, int K) {
    __shared__ __align__(16) bf16 As[BM][BK];
    __shared__ __align__(16) bf16 Bs[BN][BK];   // stored transposed: Bs[n][k]
    const int tid = threadIdx.x;
    const int lane = tid & 31, wid = tid >> 5;
    const int khalf = lane >> 4, l16 = lane & 15;
    const int m0 = blockIdx.x * BM;
    const int n0 = blockIdx.y * BN;

    const v8f vz = {0.f, 0.f, 0.f, 0.f, 0.f, 0.f, 0.f, 0.f};
    v8f acc[4] = {vz, vz, vz, vz};

    const int rA = tid >> 1;
    const int kkA = (tid & 1) * 16;
    int mA = m0 + rA;
    if (REV) { int t = mA & (LL - 1); mA = (mA - t) + (LL - 1 - t); }
    const bf16* Aptr = A + (size_t)mA * K + kkA;

    const int nnB = tid & 63;
    const int kkB = (tid >> 6) * 8;

    for (int k0 = 0; k0 < K; k0 += BK) {
        // B tile: gather-transpose through VGPRs
        __align__(16) bf16 tb[8];
#pragma unroll
        for (int i = 0; i < 8; ++i)
            tb[i] = B[(size_t)(k0 + kkB + i) * Ntot + n0 + nnB];
        // A tile: async DMA straight into LDS (ASYNCcnt path)
        async_copy_b128(Aptr + k0,     &As[rA][kkA]);
        async_copy_b128(Aptr + k0 + 8, &As[rA][kkA + 8]);
        if (k0 + BK < K) __builtin_prefetch(Aptr + k0 + BK, 0, 1);
        *reinterpret_cast<uint4*>(&Bs[nnB][kkB]) = *reinterpret_cast<const uint4*>(tb);
        wait_async0();
        __syncthreads();

        // preload all fragments, then issue WMMAs back-to-back
        v16bf af = frag16(&As[wid * 16 + l16][0], khalf);
        v16bf b0 = frag16(&Bs[0 * 16 + l16][0], khalf);
        v16bf b1 = frag16(&Bs[1 * 16 + l16][0], khalf);
        v16bf b2 = frag16(&Bs[2 * 16 + l16][0], khalf);
        v16bf b3 = frag16(&Bs[3 * 16 + l16][0], khalf);
        acc[0] = __builtin_amdgcn_wmma_f32_16x16x32_bf16(false, af, false, b0, (short)0, acc[0], false, false);
        acc[1] = __builtin_amdgcn_wmma_f32_16x16x32_bf16(false, af, false, b1, (short)0, acc[1], false, false);
        acc[2] = __builtin_amdgcn_wmma_f32_16x16x32_bf16(false, af, false, b2, (short)0, acc[2], false, false);
        acc[3] = __builtin_amdgcn_wmma_f32_16x16x32_bf16(false, af, false, b3, (short)0, acc[3], false, false);
        __syncthreads();
    }

    const int rowBase = m0 + wid * 16 + khalf * 8;   // D: VGPR r -> M = r + 8*khalf
#pragma unroll
    for (int tn = 0; tn < 4; ++tn) {
        int col = n0 + tn * 16 + l16;
#pragma unroll
        for (int r = 0; r < 8; ++r) {
            int row = rowBase + r;
            float v = acc[tn][r];
            if constexpr (EPI == 0) {
                ((bf16*)C)[(size_t)row * Ntot + col] = (bf16)v;
            } else if constexpr (EPI == 1) {
                v += bias[col];
                ((bf16*)C)[(size_t)row * Ntot + col] = (bf16)gelu_exact(v);
            } else {
                v += bias[col] + res[(size_t)row * Ntot + col];
                ((float*)C)[(size_t)row * Ntot + col] = v;
            }
        }
    }
}

// fused y = h1@W1 + rev(h2)@W2 + residual
__global__ __launch_bounds__(256) void gemm_dual_kernel(const bf16* __restrict__ A1,
                                                        const bf16* __restrict__ B1,
                                                        const bf16* __restrict__ A2,
                                                        const bf16* __restrict__ B2,
                                                        const float* __restrict__ res,
                                                        float* __restrict__ C,
                                                        int M, int Ntot, int K) {
    __shared__ __align__(16) bf16 As1[BM][BK];
    __shared__ __align__(16) bf16 As2[BM][BK];
    __shared__ __align__(16) bf16 Bs1[BN][BK];
    __shared__ __align__(16) bf16 Bs2[BN][BK];
    const int tid = threadIdx.x;
    const int lane = tid & 31, wid = tid >> 5;
    const int khalf = lane >> 4, l16 = lane & 15;
    const int m0 = blockIdx.x * BM, n0 = blockIdx.y * BN;

    const v8f vz = {0.f, 0.f, 0.f, 0.f, 0.f, 0.f, 0.f, 0.f};
    v8f acc[4] = {vz, vz, vz, vz};

    const int rA = tid >> 1, kkA = (tid & 1) * 16;
    int m1 = m0 + rA;
    int tt = m1 & (LL - 1);
    int m2 = (m1 - tt) + (LL - 1 - tt);                 // reversed sequence row
    const bf16* A1p = A1 + (size_t)m1 * K + kkA;
    const bf16* A2p = A2 + (size_t)m2 * K + kkA;
    const int nnB = tid & 63, kkB = (tid >> 6) * 8;

    for (int k0 = 0; k0 < K; k0 += BK) {
        __align__(16) bf16 t1[8], t2[8];
#pragma unroll
        for (int i = 0; i < 8; ++i) {
            t1[i] = B1[(size_t)(k0 + kkB + i) * Ntot + n0 + nnB];
            t2[i] = B2[(size_t)(k0 + kkB + i) * Ntot + n0 + nnB];
        }
        async_copy_b128(A1p + k0,     &As1[rA][kkA]);
        async_copy_b128(A1p + k0 + 8, &As1[rA][kkA + 8]);
        async_copy_b128(A2p + k0,     &As2[rA][kkA]);
        async_copy_b128(A2p + k0 + 8, &As2[rA][kkA + 8]);
        if (k0 + BK < K) {
            __builtin_prefetch(A1p + k0 + BK, 0, 1);
            __builtin_prefetch(A2p + k0 + BK, 0, 1);
        }
        *reinterpret_cast<uint4*>(&Bs1[nnB][kkB]) = *reinterpret_cast<const uint4*>(t1);
        *reinterpret_cast<uint4*>(&Bs2[nnB][kkB]) = *reinterpret_cast<const uint4*>(t2);
        wait_async0();
        __syncthreads();

        v16bf af1 = frag16(&As1[wid * 16 + l16][0], khalf);
        v16bf b10 = frag16(&Bs1[0 * 16 + l16][0], khalf);
        v16bf b11 = frag16(&Bs1[1 * 16 + l16][0], khalf);
        v16bf b12 = frag16(&Bs1[2 * 16 + l16][0], khalf);
        v16bf b13 = frag16(&Bs1[3 * 16 + l16][0], khalf);
        acc[0] = __builtin_amdgcn_wmma_f32_16x16x32_bf16(false, af1, false, b10, (short)0, acc[0], false, false);
        acc[1] = __builtin_amdgcn_wmma_f32_16x16x32_bf16(false, af1, false, b11, (short)0, acc[1], false, false);
        acc[2] = __builtin_amdgcn_wmma_f32_16x16x32_bf16(false, af1, false, b12, (short)0, acc[2], false, false);
        acc[3] = __builtin_amdgcn_wmma_f32_16x16x32_bf16(false, af1, false, b13, (short)0, acc[3], false, false);
        v16bf af2 = frag16(&As2[wid * 16 + l16][0], khalf);
        v16bf b20 = frag16(&Bs2[0 * 16 + l16][0], khalf);
        v16bf b21 = frag16(&Bs2[1 * 16 + l16][0], khalf);
        v16bf b22 = frag16(&Bs2[2 * 16 + l16][0], khalf);
        v16bf b23 = frag16(&Bs2[3 * 16 + l16][0], khalf);
        acc[0] = __builtin_amdgcn_wmma_f32_16x16x32_bf16(false, af2, false, b20, (short)0, acc[0], false, false);
        acc[1] = __builtin_amdgcn_wmma_f32_16x16x32_bf16(false, af2, false, b21, (short)0, acc[1], false, false);
        acc[2] = __builtin_amdgcn_wmma_f32_16x16x32_bf16(false, af2, false, b22, (short)0, acc[2], false, false);
        acc[3] = __builtin_amdgcn_wmma_f32_16x16x32_bf16(false, af2, false, b23, (short)0, acc[3], false, false);
        __syncthreads();
    }

    const int rowBase = m0 + wid * 16 + khalf * 8;
#pragma unroll
    for (int tn = 0; tn < 4; ++tn) {
        int col = n0 + tn * 16 + l16;
#pragma unroll
        for (int r = 0; r < 8; ++r) {
            int row = rowBase + r;
            C[(size_t)row * Ntot + col] = acc[tn][r] + res[(size_t)row * Ntot + col];
        }
    }
}

// ---------------- launcher ----------------
extern "C" void kernel_launch(void* const* d_in, const int* in_sizes, int n_in,
                              void* d_out, int out_size, void* d_ws, size_t ws_size,
                              hipStream_t stream) {
    (void)in_sizes; (void)n_in; (void)out_size; (void)ws_size;
    const float* x      = (const float*)d_in[0];
    const float* gamma1 = (const float*)d_in[1];
    const float* beta1  = (const float*)d_in[2];
    const float* dwc_w  = (const float*)d_in[3];
    const float* dwc_b  = (const float*)d_in[4];
    const float* gru1_w = (const float*)d_in[5];
    const float* gru1_o = (const float*)d_in[6];
    const float* gru2_w = (const float*)d_in[7];
    const float* gru2_o = (const float*)d_in[8];
    const float* gamma2 = (const float*)d_in[9];
    const float* beta2  = (const float*)d_in[10];
    const float* p1_w   = (const float*)d_in[11];
    const float* p1_b   = (const float*)d_in[12];
    const float* p2_w   = (const float*)d_in[13];
    const float* p2_b   = (const float*)d_in[14];

    char* wsp = (char*)d_ws;
    auto alloc = [&](size_t bytes) -> char* {
        char* p = wsp; wsp += (bytes + 255) & ~(size_t)255; return p;
    };
    float* xn  = (float*)alloc(sizeof(float) * (size_t)MTOT * DIM_);
    bf16*  xs  = (bf16*) alloc(sizeof(bf16)  * (size_t)MTOT * DIM_);
    bf16*  hg1 = (bf16*) alloc(sizeof(bf16)  * (size_t)MTOT * HGC);
    bf16*  hg2 = (bf16*) alloc(sizeof(bf16)  * (size_t)MTOT * HGC);
    bf16*  h1  = (bf16*) alloc(sizeof(bf16)  * (size_t)MTOT * DI_);
    bf16*  h2  = (bf16*) alloc(sizeof(bf16)  * (size_t)MTOT * DI_);
    float* y   = (float*)alloc(sizeof(float) * (size_t)MTOT * DIM_);
    bf16*  ynb = (bf16*) alloc(sizeof(bf16)  * (size_t)MTOT * DIM_);
    bf16*  yh  = (bf16*) alloc(sizeof(bf16)  * (size_t)MTOT * MLP_);
    bf16*  wg1 = (bf16*) alloc(sizeof(bf16)  * DIM_ * HGC);
    bf16*  wg2 = (bf16*) alloc(sizeof(bf16)  * DIM_ * HGC);
    bf16*  wo1 = (bf16*) alloc(sizeof(bf16)  * DI_ * DIM_);
    bf16*  wo2 = (bf16*) alloc(sizeof(bf16)  * DI_ * DIM_);
    bf16*  wp1 = (bf16*) alloc(sizeof(bf16)  * DIM_ * MLP_);
    bf16*  wp2 = (bf16*) alloc(sizeof(bf16)  * MLP_ * DIM_);

    auto cvt = [&](const float* s, bf16* d, int n) {
        cvt_bf16_kernel<<<(n + 255) / 256, 256, 0, stream>>>(s, d, n);
    };
    cvt(gru1_w, wg1, DIM_ * HGC);
    cvt(gru2_w, wg2, DIM_ * HGC);
    cvt(gru1_o, wo1, DI_ * DIM_);
    cvt(gru2_o, wo2, DI_ * DIM_);
    cvt(p1_w,   wp1, DIM_ * MLP_);
    cvt(p2_w,   wp2, MLP_ * DIM_);

    // 1) LayerNorm1 (fp32 out for conv precision)
    ln_kernel<float><<<MTOT / 8, 256, 0, stream>>>(x, gamma1, beta1, xn);
    // 2) depthwise 3x3 + bias -> bf16
    dwconv_kernel<<<(MTOT * DIM_) / 256, 256, 0, stream>>>(xn, dwc_w, dwc_b, xs);
    // 3) GRU input GEMMs (backward GRU = reversed A rows)
    gemm_kernel<0, false><<<dim3(MTOT / BM, HGC / BN), 256, 0, stream>>>(
        xs, wg1, nullptr, nullptr, hg1, MTOT, HGC, DIM_);
    gemm_kernel<0, true><<<dim3(MTOT / BM, HGC / BN), 256, 0, stream>>>(
        xs, wg2, nullptr, nullptr, hg2, MTOT, HGC, DIM_);
    // 4) log-space scans
    scan_kernel<<<(NB * DI_) / 256, 256, 0, stream>>>(hg1, h1);
    scan_kernel<<<(NB * DI_) / 256, 256, 0, stream>>>(hg2, h2);
    // 5) y = h1@out1 + rev(h2)@out2 + x
    gemm_dual_kernel<<<dim3(MTOT / BM, DIM_ / BN), 256, 0, stream>>>(
        h1, wo1, h2, wo2, x, y, MTOT, DIM_, DI_);
    // 6) LayerNorm2 -> bf16
    ln_kernel<bf16><<<MTOT / 8, 256, 0, stream>>>(y, gamma2, beta2, ynb);
    // 7) MLP
    gemm_kernel<1, false><<<dim3(MTOT / BM, MLP_ / BN), 256, 0, stream>>>(
        ynb, wp1, p1_b, nullptr, yh, MTOT, MLP_, DIM_);
    gemm_kernel<2, false><<<dim3(MTOT / BM, DIM_ / BN), 256, 0, stream>>>(
        yh, wp2, p2_b, y, d_out, MTOT, DIM_, MLP_);
}
